// QwQAttentionModel_64690797413047
// MI455X (gfx1250) — matrix operand order repaired and checked by reading, not compiled, not run on hardware
//
#include <hip/hip_runtime.h>
#include <hip/hip_bf16.h>
#include <math.h>
#include <stdint.h>

// ---------------------------------------------------------------------------
// Model dims (from reference)
// ---------------------------------------------------------------------------
#define DMODEL 2048
#define NH     16
#define NKH    4
#define HDIM   128
#define NL     2
#define SEQ    2048
#define GRP    (NH / NKH)          // 4 query heads per kv head
#define HHD    (NH * HDIM)         // 2048
#define KHHD   (NKH * HDIM)        // 512

typedef __bf16 bf16_t;
typedef __attribute__((ext_vector_type(16))) __bf16 v16bf;
typedef __attribute__((ext_vector_type(8)))  __bf16 v8bf;
typedef __attribute__((ext_vector_type(8)))  float  v8f;
typedef unsigned int u32x4 __attribute__((ext_vector_type(4)));
typedef int          i32x4 __attribute__((ext_vector_type(4)));
typedef int          i32x8 __attribute__((ext_vector_type(8)));

// ---------------------------------------------------------------------------
// Tensor Data Mover: DMA a 2D fp32 tile (tile_w contiguous elems x tile_h rows,
// row stride = stride_elems) from global memory into LDS at byte offset lds_off.
// Builds the D# descriptor per CDNA5 ISA 8.3/8.4 (data_size=4B, no pad/iterate,
// no multicast) and issues TENSOR_LOAD_TO_LDS. Caller waits on TENSORcnt.
// ---------------------------------------------------------------------------
__device__ __forceinline__ void tdm_load_2d_f32(uint32_t lds_off, const float* gaddr,
                                                uint32_t tile_w, uint32_t tile_h,
                                                uint64_t stride_elems)
{
    const uint64_t ga = (uint64_t)(uintptr_t)gaddr;
    u32x4 g0;
    g0[0] = 1u;                                              // count=1, user mode
    g0[1] = lds_off;                                         // lds_addr (bytes)
    g0[2] = (uint32_t)ga;                                    // global_addr[31:0]
    g0[3] = (uint32_t)((ga >> 32) & 0x01FFFFFFu) | (2u << 30); // addr[56:32] | type=2
    i32x8 g1;
    g1[0] = (int)(2u << 16);                                 // data_size=2 (4 bytes)
    g1[1] = (int)((tile_w & 0xFFFFu) << 16);                 // tensor_dim0[15:0]
    g1[2] = (int)((tile_w >> 16) | ((tile_h & 0xFFFFu) << 16)); // dim0 hi | dim1 lo
    g1[3] = (int)((tile_h >> 16) | (tile_w << 16));          // dim1 hi | tile_dim0
    g1[4] = (int)(tile_h & 0xFFFFu);                         // tile_dim1 (tile_dim2=0)
    g1[5] = (int)(uint32_t)stride_elems;                     // tensor_dim0_stride lo
    g1[6] = (int)((uint32_t)(stride_elems >> 32) & 0xFFFFu); // stride hi (dim1_stride=0)
    g1[7] = 0;
    i32x4 z4 = {0, 0, 0, 0};
#if defined(__clang_major__) && (__clang_major__ >= 23)
    i32x8 z8 = {0, 0, 0, 0, 0, 0, 0, 0};
    __builtin_amdgcn_tensor_load_to_lds(g0, g1, z4, z4, z8, 0);
#else
    __builtin_amdgcn_tensor_load_to_lds(g0, g1, z4, z4, 0);
#endif
}

// ---------------------------------------------------------------------------
// bf16-WMMA GEMM with TDM staging:  C[M,N] (+)= alpha * A[M,K] @ op(B)
//   TRANS_B=0 : op(B) = B[K,N] row-major, ldb
//   TRANS_B=1 : op(B) = B[N,K]^T row-major source, ldb
// Block = 64M x 128N (8 waves: 4M x 2N, each wave 16M x 64N), K-step 32.
// Pipeline per K-step: TDM fp32 tiles -> LDS, repack to bf16 fragment layout,
// ds_load_b128 fragments, 4x v_wmma_f32_16x16x32_bf16.
// All dims used are exact multiples of tiles; no early exits (barrier-safe).
// ---------------------------------------------------------------------------
#define AP 40   // Abf row pitch in bf16 (32 + 8 pad) -> 80B rows
#define BP 40   // Bbf row pitch in bf16

template<bool TRANS_B, bool ACCUM>
__global__ __launch_bounds__(256)
void gemm_bf16_wmma(const float* __restrict__ A, int lda,
                    const float* __restrict__ B, int ldb,
                    float* __restrict__ C, int ldc,
                    int M, int N, int K, float alpha)
{
    __shared__ __align__(16) float  Araw[64 * 32];    // 8 KB  fp32 A tile
    __shared__ __align__(16) float  Braw[32 * 128];   // 16 KB fp32 B tile
    __shared__ __align__(16) __bf16 Abf[64 * AP];     // 5 KB  bf16 [m][k]
    __shared__ __align__(16) __bf16 Bbf[128 * BP];    // 10 KB bf16 [n][k]

    const int tid   = threadIdx.x;
    const int lane  = tid & 31;
    const int wave  = tid >> 5;
    const int waveM = wave & 3;        // 0..3
    const int waveN = wave >> 2;       // 0..1
    const int half  = lane >> 4;       // 0: lanes 0-15, 1: lanes 16-31
    const int l16   = lane & 15;
    const int mBlock = blockIdx.y * 64;
    const int nBlock = blockIdx.x * 128;

    const uint32_t a_lds = (uint32_t)(uintptr_t)&Araw[0];
    const uint32_t b_lds = (uint32_t)(uintptr_t)&Braw[0];

    v8f acc[4] = {};

    for (int kk = 0; kk < K; kk += 32) {
        // ---- stage 1: TDM DMA of fp32 tiles into LDS (wave 0 only) ----
        if (wave == 0) {
            tdm_load_2d_f32(a_lds, A + (size_t)mBlock * lda + kk,
                            32u, 64u, (uint64_t)lda);
            if (TRANS_B)
                tdm_load_2d_f32(b_lds, B + (size_t)nBlock * ldb + kk,
                                32u, 128u, (uint64_t)ldb);
            else
                tdm_load_2d_f32(b_lds, B + (size_t)kk * ldb + nBlock,
                                128u, 32u, (uint64_t)ldb);
            __builtin_amdgcn_s_wait_tensorcnt(0);
        }
        __syncthreads();

        // ---- stage 2: repack fp32 -> bf16 fragment layout (all threads) ----
        {   // A: Abf[m][k] = Araw[m][k]; 8 elems/thread, b128 LDS traffic
            const int m  = tid >> 2;
            const int k0 = (tid & 3) * 8;
            v8bf r;
            #pragma unroll
            for (int j = 0; j < 8; ++j) r[j] = (bf16_t)Araw[m * 32 + k0 + j];
            *(v8bf*)&Abf[m * AP + k0] = r;
        }
        if (TRANS_B) {  // Braw is [n:128][k:32] -> straight copy
            const int n  = tid >> 1;
            const int k0 = (tid & 1) * 16;
            v8bf lo, hi;
            #pragma unroll
            for (int j = 0; j < 8; ++j) lo[j] = (bf16_t)Braw[n * 32 + k0 + j];
            #pragma unroll
            for (int j = 0; j < 8; ++j) hi[j] = (bf16_t)Braw[n * 32 + k0 + 8 + j];
            *(v8bf*)&Bbf[n * BP + k0]     = lo;
            *(v8bf*)&Bbf[n * BP + k0 + 8] = hi;
        } else {        // Braw is [k:32][n:128] -> transpose in LDS
            const int n  = tid & 127;
            const int k0 = (tid >> 7) * 16;
            v8bf lo, hi;
            #pragma unroll
            for (int j = 0; j < 8; ++j) lo[j] = (bf16_t)Braw[(k0 + j) * 128 + n];
            #pragma unroll
            for (int j = 0; j < 8; ++j) hi[j] = (bf16_t)Braw[(k0 + 8 + j) * 128 + n];
            *(v8bf*)&Bbf[n * BP + k0]     = lo;
            *(v8bf*)&Bbf[n * BP + k0 + 8] = hi;
        }
        __syncthreads();

        // ---- stage 3: fragments (2x ds_load_b128 each) + 4 WMMAs ----
        // A frag here uses contiguous K (same order as B): lanes 0-15 elems = K 0-15, lanes 16-31 = K 16-31
        const int am  = waveM * 16 + l16;
        const int ak0 = half ? 16 : 0;
        const int ak1 = half ? 24 : 8;
        v8bf alo = *(const v8bf*)&Abf[am * AP + ak0];
        v8bf ahi = *(const v8bf*)&Abf[am * AP + ak1];
        v16bf afrag = __builtin_shufflevector(alo, ahi,
            0,1,2,3,4,5,6,7,8,9,10,11,12,13,14,15);

        // B frag here uses contiguous K (same order as A): lanes 0-15 hold K 0-15, lanes 16-31 hold K 16-31
        const int bk = half ? 16 : 0;
        #pragma unroll
        for (int sub = 0; sub < 4; ++sub) {
            const int n = waveN * 64 + sub * 16 + l16;
            v8bf blo = *(const v8bf*)&Bbf[n * BP + bk];
            v8bf bhi = *(const v8bf*)&Bbf[n * BP + bk + 8];
            v16bf bfrag = __builtin_shufflevector(blo, bhi,
                0,1,2,3,4,5,6,7,8,9,10,11,12,13,14,15);
            acc[sub] = __builtin_amdgcn_wmma_f32_16x16x32_bf16(
                false, afrag, false, bfrag, (short)0, acc[sub], false, false);
        }
        __syncthreads();   // protect Abf/Bbf (and Araw/Braw) for next K-step
    }

    // ---- store (f32 C/D layout: VGPR r -> M = base + r + 8*half, N = l16) ----
    #pragma unroll
    for (int sub = 0; sub < 4; ++sub) {
        const int n = nBlock + waveN * 64 + sub * 16 + l16;
        #pragma unroll
        for (int r = 0; r < 8; ++r) {
            const int m = mBlock + waveM * 16 + r + (half ? 8 : 0);
            const size_t idx = (size_t)m * ldc + n;
            const float val = alpha * acc[sub][r];
            if (ACCUM) C[idx] += val; else C[idx] = val;
        }
    }
}

template<bool TRANS_B, bool ACCUM>
static inline void launch_gemm(const float* A, int lda, const float* B, int ldb,
                               float* C, int ldc, int M, int N, int K,
                               float alpha, hipStream_t stream)
{
    dim3 grid(N / 128, M / 64);
    gemm_bf16_wmma<TRANS_B, ACCUM><<<grid, 256, 0, stream>>>(
        A, lda, B, ldb, C, ldc, M, N, K, alpha);
}

// ---------------------------------------------------------------------------
// Embedding gather: one block per token
// ---------------------------------------------------------------------------
__global__ __launch_bounds__(256)
void embed_kernel(const int* __restrict__ ids, const float* __restrict__ emb,
                  float* __restrict__ x, int d)
{
    const int tok = blockIdx.x;
    const int id  = ids[tok];
    const float* src = emb + (size_t)id * d;
    float* dst = x + (size_t)tok * d;
    for (int i = threadIdx.x; i < d; i += blockDim.x) dst[i] = src[i];
}

// ---------------------------------------------------------------------------
// LayerNorm: one block (256 thr) per row of D
// ---------------------------------------------------------------------------
__global__ __launch_bounds__(256)
void layernorm_kernel(const float* __restrict__ x, const float* __restrict__ g,
                      const float* __restrict__ b, float* __restrict__ y, int d)
{
    __shared__ float red[256];
    const int row = blockIdx.x;
    const float* xr = x + (size_t)row * d;

    float s = 0.f;
    for (int i = threadIdx.x; i < d; i += blockDim.x) s += xr[i];
    red[threadIdx.x] = s; __syncthreads();
    for (int w = 128; w > 0; w >>= 1) {
        if (threadIdx.x < w) red[threadIdx.x] += red[threadIdx.x + w];
        __syncthreads();
    }
    const float mu = red[0] / d;
    __syncthreads();

    float vs = 0.f;
    for (int i = threadIdx.x; i < d; i += blockDim.x) {
        const float t = xr[i] - mu; vs += t * t;
    }
    red[threadIdx.x] = vs; __syncthreads();
    for (int w = 128; w > 0; w >>= 1) {
        if (threadIdx.x < w) red[threadIdx.x] += red[threadIdx.x + w];
        __syncthreads();
    }
    const float rstd = rsqrtf(red[0] / d + 1e-5f);
    __syncthreads();

    float* yr = y + (size_t)row * d;
    for (int i = threadIdx.x; i < d; i += blockDim.x)
        yr[i] = (xr[i] - mu) * rstd * g[i] + b[i];
}

// ---------------------------------------------------------------------------
// Row softmax in place: one block per row of n
// ---------------------------------------------------------------------------
__global__ __launch_bounds__(256)
void softmax_kernel(float* __restrict__ p, int n)
{
    __shared__ float red[256];
    float* row = p + (size_t)blockIdx.x * n;

    float mx = -INFINITY;
    for (int i = threadIdx.x; i < n; i += blockDim.x) mx = fmaxf(mx, row[i]);
    red[threadIdx.x] = mx; __syncthreads();
    for (int w = 128; w > 0; w >>= 1) {
        if (threadIdx.x < w) red[threadIdx.x] = fmaxf(red[threadIdx.x], red[threadIdx.x + w]);
        __syncthreads();
    }
    mx = red[0]; __syncthreads();

    float s = 0.f;
    for (int i = threadIdx.x; i < n; i += blockDim.x) {
        const float e = expf(row[i] - mx); row[i] = e; s += e;
    }
    red[threadIdx.x] = s; __syncthreads();
    for (int w = 128; w > 0; w >>= 1) {
        if (threadIdx.x < w) red[threadIdx.x] += red[threadIdx.x + w];
        __syncthreads();
    }
    const float inv = 1.0f / red[0];
    __syncthreads();
    for (int i = threadIdx.x; i < n; i += blockDim.x) row[i] *= inv;
}

// ---------------------------------------------------------------------------
// Launcher
// ---------------------------------------------------------------------------
extern "C" void kernel_launch(void* const* d_in, const int* in_sizes, int n_in,
                              void* d_out, int out_size, void* d_ws, size_t ws_size,
                              hipStream_t stream)
{
    (void)in_sizes; (void)n_in; (void)out_size; (void)ws_size;

    const int*   input_ids = (const int*)  d_in[0];
    const float* emb       = (const float*)d_in[1];
    const float* ln_g      = (const float*)d_in[2];
    const float* ln_b      = (const float*)d_in[3];
    const float* wq        = (const float*)d_in[4];
    const float* wk        = (const float*)d_in[5];
    const float* wv        = (const float*)d_in[6];
    const float* wo        = (const float*)d_in[7];
    const float* norm_g    = (const float*)d_in[8];
    const float* norm_b    = (const float*)d_in[9];

    float* out   = (float*)d_out;
    float* x_out = out;                         // [S, D]
    float* attns = out + (size_t)SEQ * DMODEL;  // [L, H, S, S]

    // workspace carve-up (~75.5 MB of fp32)
    float* ws = (float*)d_ws;
    float* x    = ws; ws += (size_t)SEQ * DMODEL;   // residual stream
    float* h    = ws; ws += (size_t)SEQ * DMODEL;   // LN output
    float* q    = ws; ws += (size_t)SEQ * HHD;      // [S, H*HD]
    float* kbuf = ws; ws += (size_t)SEQ * KHHD;     // [S, KH*HD]
    float* vbuf = ws; ws += (size_t)SEQ * KHHD;     // [S, KH*HD]
    float* o    = ws; ws += (size_t)SEQ * HHD;      // attn output [S, H*HD]

    const float scale = 0.08838834764831845f;       // 1/sqrt(128)

    // x = embed[input_ids]
    embed_kernel<<<SEQ, 256, 0, stream>>>(input_ids, emb, x, DMODEL);

    for (int l = 0; l < NL; ++l) {
        // h = LN(x) * g + b
        layernorm_kernel<<<SEQ, 256, 0, stream>>>(
            x, ln_g + (size_t)l * DMODEL, ln_b + (size_t)l * DMODEL, h, DMODEL);

        // projections
        launch_gemm<false, false>(h, DMODEL, wq + (size_t)l * DMODEL * HHD,  HHD,
                                  q,    HHD,  SEQ, HHD,  DMODEL, 1.0f, stream);
        launch_gemm<false, false>(h, DMODEL, wk + (size_t)l * DMODEL * KHHD, KHHD,
                                  kbuf, KHHD, SEQ, KHHD, DMODEL, 1.0f, stream);
        launch_gemm<false, false>(h, DMODEL, wv + (size_t)l * DMODEL * KHHD, KHHD,
                                  vbuf, KHHD, SEQ, KHHD, DMODEL, 1.0f, stream);

        float* attns_l = attns + (size_t)l * NH * SEQ * SEQ;

        // scores = scale * q_h @ k_kh^T, written straight into the output slice
        for (int hh = 0; hh < NH; ++hh) {
            const int kh = hh / GRP;
            float* probs = attns_l + (size_t)hh * SEQ * SEQ;
            launch_gemm<true, false>(q + (size_t)hh * HDIM, HHD,
                                     kbuf + (size_t)kh * HDIM, KHHD,
                                     probs, SEQ, SEQ, SEQ, HDIM, scale, stream);
        }

        // softmax over last dim for all H*S rows of this layer
        softmax_kernel<<<NH * SEQ, 256, 0, stream>>>(attns_l, SEQ);

        // o_h = probs_h @ v_kh  (head-strided output columns of o)
        for (int hh = 0; hh < NH; ++hh) {
            const int kh = hh / GRP;
            float* probs = attns_l + (size_t)hh * SEQ * SEQ;
            launch_gemm<false, false>(probs, SEQ,
                                      vbuf + (size_t)kh * HDIM, KHHD,
                                      o + (size_t)hh * HDIM, HHD,
                                      SEQ, HDIM, SEQ, 1.0f, stream);
        }

        // x += o @ wo[l]
        launch_gemm<false, true>(o, HHD, wo + (size_t)l * HHD * DMODEL, DMODEL,
                                 x, DMODEL, SEQ, DMODEL, HHD, 1.0f, stream);
    }

    // final LayerNorm -> first S*D floats of d_out
    layernorm_kernel<<<SEQ, 256, 0, stream>>>(x, norm_g, norm_b, x_out, DMODEL);
}
